// RGCNModel_3985729651460
// MI455X (gfx1250) — matrix operand order repaired and checked by reading, not compiled, not run on hardware
//
#include <hip/hip_runtime.h>
#include <hip/hip_bf16.h>

// RGCN 2-layer forward for MI455X (gfx1250, wave32, WMMA).
//   1. cnt[(dst,etype)] edge counts (int atomics, reused by both layers).
//   2. Convert x -> bf16; transpose+convert W1/W2/root1/root2 -> bf16 col-major.
//   3. Per layer: xW[N,R,128] via v_wmma_f32_16x16x32_bf16 (8 waves = 8 relations
//      per 16-node tile). A sched_barrier between a k-chunk's 18 16-byte loads
//      and its 8 WMMAs keeps the loads batched (16 in flight, one wait) instead
//      of the scheduler sinking them to one clause per WMMA.
//      out = x@root + b via the same WMMA path; per-edge scatter adds
//      xW[src,etype,:] * (1/cnt) into out (f32 global atomics) — reproducing
//      segment-mean + sum-over-relations exactly.
//   4. ReLU + bf16 convert between layers.
// xW (410 MB) exceeds the 192 MB L2 -> non-temporal stores/loads for it so L2
// stays hot for weights (0.5 MB) and the atomic destination rows.

#define NNODES 100000
#define NEDGES 640000
#define DIM    128
#define NREL   8

typedef __bf16 bf16_t;
typedef __attribute__((ext_vector_type(16))) __bf16 v16bf;
typedef __attribute__((ext_vector_type(8)))  __bf16 v8bf;
typedef __attribute__((ext_vector_type(8)))  float  v8f;
typedef __attribute__((ext_vector_type(4)))  float  v4f;

// ---------------- elementwise / setup kernels ----------------

__global__ void k_zero_i32(int* __restrict__ p, int n) {
  int g = blockIdx.x * blockDim.x + threadIdx.x;
  if (g < n) p[g] = 0;
}

__global__ void k_count(const int* __restrict__ dst, const int* __restrict__ et,
                        int* __restrict__ cnt, int nE) {
  int g = blockIdx.x * blockDim.x + threadIdx.x;
  if (g < nE) atomicAdd(&cnt[dst[g] * NREL + et[g]], 1);
}

__global__ void k_f32_to_bf16(const float* __restrict__ s, bf16_t* __restrict__ d, int n) {
  int g = blockIdx.x * blockDim.x + threadIdx.x;
  if (g < n) d[g] = (bf16_t)s[g];
}

// W[m][d][o] (row-major, contraction dim d) -> Wt[m][o][d] bf16 so that a
// WMMA B-fragment's two 8-K runs (one output column) are each contiguous.
__global__ void k_transpose_bf16(const float* __restrict__ W, bf16_t* __restrict__ Wt, int nmat) {
  int g = blockIdx.x * blockDim.x + threadIdx.x;
  int total = nmat * DIM * DIM;
  if (g >= total) return;
  int m   = g / (DIM * DIM);
  int rem = g - m * (DIM * DIM);
  int d   = rem / DIM;
  int o   = rem - d * DIM;
  Wt[(size_t)m * DIM * DIM + (size_t)o * DIM + d] = (bf16_t)W[g];
}

__global__ void k_relu_cvt(const float* __restrict__ h, bf16_t* __restrict__ hb, int n) {
  int g = blockIdx.x * blockDim.x + threadIdx.x;
  if (g < n) {
    float v = h[g];
    hb[g] = (bf16_t)(v > 0.f ? v : 0.f);
  }
}

// ---------------- WMMA fragment loaders ----------------
// A (16x32 bf16, MxK): lane holds row = lane%16; elements i=0..15 map to
//   K = kk*32 + (i/8)*16 + (lane/16)*8 + (i%8)  -> two contiguous 16B loads.
__device__ __forceinline__ v16bf load_a_frag(const bf16_t* __restrict__ arow,
                                             int kk, int laneHi) {
  const v8bf lo = *(const v8bf*)(arow + kk * 32 + laneHi * 8);
  const v8bf hi = *(const v8bf*)(arow + kk * 32 + laneHi * 8 + 16);
  v16bf a;
#pragma unroll
  for (int i = 0; i < 8; ++i) { a[i] = lo[i]; a[i + 8] = hi[i]; }
  return a;
}

// B (32x16 bf16, KxN): lane holds column = lane%16; element i maps to
//   K = kk*32 + (i/8)*16 + (lane/16)*8 + (i%8)  -> two contiguous 16B loads.
__device__ __forceinline__ v16bf load_b_frag(const bf16_t* __restrict__ bcol,
                                             int kk, int laneHi) {
  const v8bf lo = *(const v8bf*)(bcol + kk * 32 + laneHi * 8);
  const v8bf hi = *(const v8bf*)(bcol + kk * 32 + laneHi * 8 + 16);
  v16bf b;
#pragma unroll
  for (int i = 0; i < 8; ++i) { b[i] = lo[i]; b[i + 8] = hi[i]; }
  return b;
}

// ---------------- GEMM kernels ----------------
// xW[n, r, :] = x_bf16[n, :] @ W[r]  (8 waves per block = 8 relations,
// each wave: 16x128 output tile = 8 accumulators, K=128 in 4 WMMA k-chunks).
__global__ __launch_bounds__(256) void k_xw_gemm(const bf16_t* __restrict__ xb,
                                                 const bf16_t* __restrict__ Wt,
                                                 float* __restrict__ xW) {
  const int wave   = threadIdx.x >> 5;   // relation id
  const int lane   = threadIdx.x & 31;
  const int laneHi = lane >> 4;
  const int laneLo = lane & 15;
  const int node0  = blockIdx.x * 16;

  const bf16_t* arow = xb + (size_t)(node0 + laneLo) * DIM;
  const bf16_t* wmat = Wt + (size_t)wave * DIM * DIM;
  const bf16_t* bcol = wmat + (size_t)laneLo * DIM;

  v8f acc[8];
#pragma unroll
  for (int nt = 0; nt < 8; ++nt)
#pragma unroll
    for (int i = 0; i < 8; ++i) acc[nt][i] = 0.f;

#pragma unroll
  for (int kk = 0; kk < 4; ++kk) {
    if (kk < 3) __builtin_prefetch(arow + (kk + 1) * 32, 0, 3);
    v16bf a = load_a_frag(arow, kk, laneHi);
    v16bf bfrag[8];
#pragma unroll
    for (int nt = 0; nt < 8; ++nt)
      bfrag[nt] = load_b_frag(bcol + (size_t)(nt * 16) * DIM, kk, laneHi);
    // Keep all 18 loads of this k-chunk above the WMMA burst: one wait,
    // 8 back-to-back matrix ops, ~16 loads in flight for latency hiding.
    __builtin_amdgcn_sched_barrier(0);
#pragma unroll
    for (int nt = 0; nt < 8; ++nt)
      acc[nt] = __builtin_amdgcn_wmma_f32_16x16x32_bf16(
          false, a, false, bfrag[nt], (short)0, acc[nt], false, false);
  }

  // C/D layout: VGPR v -> row M = (lane/16)*8 + v, col N = lane%16.
  // xW is streamed (410 MB > L2): non-temporal stores.
#pragma unroll
  for (int nt = 0; nt < 8; ++nt)
#pragma unroll
    for (int v = 0; v < 8; ++v) {
      int m = laneHi * 8 + v;
      __builtin_nontemporal_store(
          acc[nt][v],
          xW + (size_t)(node0 + m) * (NREL * DIM) + (size_t)wave * DIM + nt * 16 + laneLo);
    }
}

// out[n, :] = x_bf16[n, :] @ root + bias  (8 waves = 8 column tiles of 16).
__global__ __launch_bounds__(256) void k_root_gemm(const bf16_t* __restrict__ xb,
                                                   const bf16_t* __restrict__ rootT,
                                                   const float* __restrict__ bias,
                                                   float* __restrict__ out) {
  const int wave   = threadIdx.x >> 5;   // output column tile
  const int lane   = threadIdx.x & 31;
  const int laneHi = lane >> 4;
  const int laneLo = lane & 15;
  const int node0  = blockIdx.x * 16;
  const int col    = wave * 16 + laneLo;

  const bf16_t* arow = xb + (size_t)(node0 + laneLo) * DIM;
  const bf16_t* bcol = rootT + (size_t)col * DIM;

  v8f acc;
#pragma unroll
  for (int i = 0; i < 8; ++i) acc[i] = 0.f;

  // Load all fragments for the whole K=128 first, then run the 4 WMMAs.
  v16bf afrag[4], bfrag[4];
#pragma unroll
  for (int kk = 0; kk < 4; ++kk) {
    afrag[kk] = load_a_frag(arow, kk, laneHi);
    bfrag[kk] = load_b_frag(bcol, kk, laneHi);
  }
  __builtin_amdgcn_sched_barrier(0);
#pragma unroll
  for (int kk = 0; kk < 4; ++kk)
    acc = __builtin_amdgcn_wmma_f32_16x16x32_bf16(
        false, afrag[kk], false, bfrag[kk], (short)0, acc, false, false);

  float bv = bias[col];
#pragma unroll
  for (int v = 0; v < 8; ++v) {
    int m = laneHi * 8 + v;
    out[(size_t)(node0 + m) * DIM + col] = acc[v] + bv;
  }
}

// ---------------- edge scatter (mean folded into per-edge scale) ----------------
// One wave per edge: lane l moves floats [4l, 4l+4) of the 128-float message as
// a float4 non-temporal load + 4 f32 global atomics. Indices loaded once per lane
// (wave-uniform), not once per float.
__global__ __launch_bounds__(256) void k_edge_scatter(const float* __restrict__ xW,
                               const int* __restrict__ src, const int* __restrict__ dst,
                               const int* __restrict__ et, const int* __restrict__ cnt,
                               float* __restrict__ out) {
  const int wave = threadIdx.x >> 5;
  const int lane = threadIdx.x & 31;
  const int e    = blockIdx.x * 8 + wave;
  if (e >= NEDGES) return;
  const int s  = src[e];
  const int dd = dst[e];
  const int t  = et[e];
  const int c  = cnt[dd * NREL + t];
  const float sc = 1.0f / (float)(c > 0 ? c : 1);

  const v4f m = __builtin_nontemporal_load(
      (const v4f*)(xW + (size_t)s * (NREL * DIM) + (size_t)t * DIM + lane * 4));
  float* o = out + (size_t)dd * DIM + lane * 4;
  atomicAdd(o + 0, m[0] * sc);
  atomicAdd(o + 1, m[1] * sc);
  atomicAdd(o + 2, m[2] * sc);
  atomicAdd(o + 3, m[3] * sc);
}

// ---------------- host launch ----------------
extern "C" void kernel_launch(void* const* d_in, const int* in_sizes, int n_in,
                              void* d_out, int out_size, void* d_ws, size_t ws_size,
                              hipStream_t stream) {
  (void)in_sizes; (void)n_in; (void)out_size; (void)ws_size;
  const float* x     = (const float*)d_in[0];
  const int*   eidx  = (const int*)d_in[1];
  const int*   etype = (const int*)d_in[2];
  const float* W1    = (const float*)d_in[3];
  const float* r1    = (const float*)d_in[4];
  const float* b1    = (const float*)d_in[5];
  const float* W2    = (const float*)d_in[6];
  const float* r2    = (const float*)d_in[7];
  const float* b2    = (const float*)d_in[8];
  const int* src = eidx;
  const int* dst = eidx + NEDGES;
  float* out = (float*)d_out;

  char* ws = (char*)d_ws;
  size_t off = 0;
  auto alloc = [&](size_t bytes) -> char* {
    char* p = ws + off;
    off += (bytes + 255) & ~(size_t)255;
    return p;
  };
  int*    cnt = (int*)   alloc(sizeof(int)    * (size_t)NNODES * NREL);
  bf16_t* xb  = (bf16_t*)alloc(sizeof(bf16_t) * (size_t)NNODES * DIM);
  bf16_t* hb  = (bf16_t*)alloc(sizeof(bf16_t) * (size_t)NNODES * DIM);
  float*  h   = (float*) alloc(sizeof(float)  * (size_t)NNODES * DIM);
  bf16_t* W1t = (bf16_t*)alloc(sizeof(bf16_t) * (size_t)NREL * DIM * DIM);
  bf16_t* W2t = (bf16_t*)alloc(sizeof(bf16_t) * (size_t)NREL * DIM * DIM);
  bf16_t* r1t = (bf16_t*)alloc(sizeof(bf16_t) * (size_t)DIM * DIM);
  bf16_t* r2t = (bf16_t*)alloc(sizeof(bf16_t) * (size_t)DIM * DIM);
  float*  xW  = (float*) alloc(sizeof(float)  * (size_t)NNODES * NREL * DIM);

  const int T = 256;
  // edge counts (shared by both layers)
  k_zero_i32<<<(NNODES * NREL + T - 1) / T, T, 0, stream>>>(cnt, NNODES * NREL);
  k_count<<<(NEDGES + T - 1) / T, T, 0, stream>>>(dst, etype, cnt, NEDGES);
  // precision/layout conversion
  k_f32_to_bf16<<<(NNODES * DIM + T - 1) / T, T, 0, stream>>>(x, xb, NNODES * DIM);
  k_transpose_bf16<<<(NREL * DIM * DIM + T - 1) / T, T, 0, stream>>>(W1, W1t, NREL);
  k_transpose_bf16<<<(NREL * DIM * DIM + T - 1) / T, T, 0, stream>>>(W2, W2t, NREL);
  k_transpose_bf16<<<(DIM * DIM + T - 1) / T, T, 0, stream>>>(r1, r1t, 1);
  k_transpose_bf16<<<(DIM * DIM + T - 1) / T, T, 0, stream>>>(r2, r2t, 1);

  const int nodeBlocks = NNODES / 16;            // 6250 (exact)
  const int edgeBlocks = NEDGES / 8;             // 80000 (one wave per edge)

  // ----- layer 1 -----
  k_xw_gemm<<<nodeBlocks, T, 0, stream>>>(xb, W1t, xW);
  k_root_gemm<<<nodeBlocks, T, 0, stream>>>(xb, r1t, b1, h);      // init h = x@root1 + b1
  k_edge_scatter<<<edgeBlocks, T, 0, stream>>>(xW, src, dst, etype, cnt, h);
  k_relu_cvt<<<(NNODES * DIM + T - 1) / T, T, 0, stream>>>(h, hb, NNODES * DIM);

  // ----- layer 2 -----
  k_xw_gemm<<<nodeBlocks, T, 0, stream>>>(hb, W2t, xW);
  k_root_gemm<<<nodeBlocks, T, 0, stream>>>(hb, r2t, b2, out);    // init out = h@root2 + b2
  k_edge_scatter<<<edgeBlocks, T, 0, stream>>>(xW, src, dst, etype, cnt, out);
}
